// BertNER_34445637714270
// MI455X (gfx1250) — matrix operand order repaired and checked by reading, not compiled, not run on hardware
//
#include <hip/hip_runtime.h>
#include <hip/hip_bf16.h>
#include <stdint.h>

#define BB 256
#define TT 512
#define KK 128
#define STARTT (KK - 2)
#define ENDD   (KK - 1)
#define NB 16   // batches per forward workgroup (N dim of WMMA tile)

typedef __attribute__((ext_vector_type(16))) _Float16 v16h;
typedef __attribute__((ext_vector_type(8)))  _Float16 v8h;
typedef __attribute__((ext_vector_type(8)))  float    v8f;
typedef __attribute__((ext_vector_type(4)))  int      v4i;

// ---- workspace layout (bytes) ----
#define WS_EXPT   0                       // _Float16 [K*K]  (32 KB)
#define WS_ROWMAX (32 * 1024)             // float [K]
#define WS_GOLD   (WS_ROWMAX + 1024)      // float [B]
#define WS_BEST   (WS_GOLD + 1024)        // int   [B]
#define WS_BP     (64 * 1024)             // uchar [T*B*K] = 16.7 MB (L2-resident)

// ---- CDNA5 async-to-LDS path (guarded; falls back to direct loads) ----
#if __has_builtin(__builtin_amdgcn_global_load_async_to_lds_b128)
#define ASYNC_LDS 1
typedef __attribute__((address_space(1))) v4i gv4i;   // global int4
typedef __attribute__((address_space(3))) v4i lv4i;   // LDS int4
__device__ __forceinline__ void async_copy16(const float* g, float* l) {
  // integer->pointer reinterpret compiles regardless of AS qualifiers;
  // AS1 value == flat global address, AS3 uses the low 32 aperture bits.
  __builtin_amdgcn_global_load_async_to_lds_b128(
      (gv4i*)(uintptr_t)g, (lv4i*)(uintptr_t)l, /*offset=*/0, /*cpol=*/0);
}
__device__ __forceinline__ void async_wait0() {
#if __has_builtin(__builtin_amdgcn_s_wait_asynccnt)
  __builtin_amdgcn_s_wait_asynccnt(0);
#else
  asm volatile("s_wait_asynccnt 0x0" ::: "memory");
#endif
}
#else
#define ASYNC_LDS 0
#endif

// ---------------------------------------------------------------------------
// Precompute expT[i][j] = exp(trans[i][j] - rowmax[i]) in f16 (values in (0,1])
// ---------------------------------------------------------------------------
__global__ void crf_prep(const float* __restrict__ trans,
                         _Float16* __restrict__ expT,
                         float* __restrict__ rowmax) {
  const int i = threadIdx.x;  // 0..127
  float m = -1e30f;
  for (int j = 0; j < KK; ++j) m = fmaxf(m, trans[i * KK + j]);
  rowmax[i] = m;
  for (int j = 0; j < KK; ++j)
    expT[i * KK + j] = (_Float16)__expf(trans[i * KK + j] - m);
}

// ---------------------------------------------------------------------------
// Gold path score: one lane per batch.
// ---------------------------------------------------------------------------
__global__ void crf_gold(const float* __restrict__ feats,
                         const float* __restrict__ trans,
                         const int* __restrict__ tags,
                         float* __restrict__ gold) {
  const int b = blockIdx.x * blockDim.x + threadIdx.x;
  if (b >= BB) return;
  float emit = 0.f, tr = 0.f;
  int prev = STARTT;
  for (int t = 0; t < TT; ++t) {
    const int tg = tags[b * TT + t];
    emit += feats[((size_t)b * TT + t) * KK + tg];
    tr   += trans[tg * KK + prev];
    prev = tg;
  }
  tr += trans[ENDD * KK + prev];
  gold[b] = emit + tr;
}

// ---------------------------------------------------------------------------
// Forward log-partition via max-normalized linear-space GEMM on WMMA.
// One workgroup (8 waves / 256 threads) owns 16 batches; alpha lives in LDS.
// Per timestep: D[i,b] = expT(f16, LDS) @ ea(f16, LDS), 4x v_wmma per wave.
// feats tiles double-buffered in LDS via async-to-LDS DMA (overlapped with t).
// ---------------------------------------------------------------------------
__launch_bounds__(256)
__global__ void crf_forward(const float* __restrict__ feats,
                            const float* __restrict__ trans,
                            const _Float16* __restrict__ expT,
                            const float* __restrict__ rowmax,
                            const float* __restrict__ gold,
                            float* __restrict__ loss_out) {
  __shared__ __align__(32) _Float16 lsA[KK][KK];      // expT[i][j]   32 KB
  __shared__ __align__(32) _Float16 lsEA[NB][KK];     // ea[b][j]      4 KB
  __shared__ float lsAlpha[NB][KK + 4];               // alpha[b][i], padded
  __shared__ float lsM[NB];                           // per-batch running max
  __shared__ float lsRM[KK];                          // rowmax of trans
#if ASYNC_LDS
  __shared__ __align__(16) float lsF[2][NB][KK + 4];  // feats double buffer
#endif

  const int tid = threadIdx.x;
  const int w   = tid >> 5;   // wave id 0..7 -> M-tile
  const int L   = tid & 31;   // lane
  const int hi  = L >> 4;     // lane half
  const int col = L & 15;     // batch column (N) / A-row offset
  const int b0  = blockIdx.x * NB;

  // stage expT -> LDS (16B chunks), rowmax -> LDS
  {
    _Float16* abase = &lsA[0][0];
    for (int idx = tid; idx < (KK * KK) / 8; idx += 256)
      *(v8h*)(abase + idx * 8) = *(const v8h*)(expT + idx * 8);
  }
  if (tid < KK) lsRM[tid] = rowmax[tid];
  // alpha_0 = -inf except START=0  =>  m=0, ea = onehot(START)
  for (int e = tid; e < NB * KK; e += 256) {
    const int bb = e >> 7, j = e & (KK - 1);
    lsEA[bb][j] = (_Float16)((j == STARTT) ? 1.0f : 0.0f);
  }
  if (tid < NB) lsM[tid] = 0.0f;

#if ASYNC_LDS
  // prologue: kick off DMA of feats tile for t=0 (512 x 16B chunks, 2/thread)
#pragma unroll
  for (int q = 0; q < 2; ++q) {
    const int c = tid + q * 256;
    const int bb = c >> 5, e = (c & 31) * 4;
    async_copy16(&feats[((size_t)(b0 + bb) * TT + 0) * KK + e], &lsF[0][bb][e]);
  }
#endif
  __syncthreads();

  const int arow  = w * 16 + col;      // A-matrix row this lane supplies
  const int ibase = w * 16 + hi * 8;   // first D row (M) this lane holds

  for (int t = 0; t < TT; ++t) {
#if ASYNC_LDS
    const int buf = t & 1;
    async_wait0();       // feats tile for t has landed (per-wave)
    __syncthreads();     // ...and for every wave's chunks
    if (t + 1 < TT) {    // kick off DMA for t+1; hidden under WMMA + finalize
#pragma unroll
      for (int q = 0; q < 2; ++q) {
        const int c = tid + q * 256;
        const int bb = c >> 5, e = (c & 31) * 4;
        async_copy16(&feats[((size_t)(b0 + bb) * TT + (t + 1)) * KK + e],
                     &lsF[buf ^ 1][bb][e]);
      }
    }
#endif

    // ---- D[i,b] = sum_j expT[i,j] * ea[j,b]  (K=128 in 4 chunks of 32) ----
    v8f acc = {};
#pragma unroll
    for (int kc = 0; kc < 4; ++kc) {
      // A in contiguous k order (this code's choice, same as B): lanes 0-15 carry K [0..15], lanes 16-31 K [16..31]
      const v8h alo = *(const v8h*)&lsA[arow][kc * 32 + hi * 16];
      const v8h ahi = *(const v8h*)&lsA[arow][kc * 32 + hi * 16 + 8];
      v16h a;
#pragma unroll
      for (int q = 0; q < 8; ++q) { a[q] = alo[q]; a[q + 8] = ahi[q]; }
      // B in the same contiguous k order as A: lane-half selects K 16-group, lane&15 selects column
      const v16h bm = *(const v16h*)&lsEA[col][kc * 32 + hi * 16];
      acc = __builtin_amdgcn_wmma_f32_16x16x32_f16(
          /*neg_a=*/false, a, /*neg_b=*/false, bm,
          /*c_mod=*/(short)0, acc, /*reuse_a=*/false, /*reuse_b=*/false);
    }

#if !ASYNC_LDS
    // fallback: prefetch next timestep's feats tile (4 lines per batch)
    if (t + 1 < TT && tid < 64) {
      const int pb = tid >> 2, chunk = tid & 3;
      __builtin_prefetch(
          &feats[((size_t)(b0 + pb) * TT + (t + 1)) * KK + chunk * 32], 0, 0);
    }
#endif

    // ---- alpha_new[i,b] = feat + rm[i] + m[b] + log(D[i,b]) ----
    const float mb = lsM[col];
#if ASYNC_LDS
    const float* fp = &lsF[buf][col][ibase];
#else
    const float* fp = &feats[((size_t)(b0 + col) * TT + t) * KK + ibase];
#endif
#pragma unroll
    for (int r = 0; r < 8; ++r) {
      const int i = ibase + r;
      lsAlpha[col][i] = fp[r] + lsRM[i] + mb + __logf(fmaxf(acc[r], 1e-30f));
    }
    __syncthreads();

    // new per-batch max
    if (tid < NB) {
      float m = -1e30f;
      for (int i = 0; i < KK; ++i) m = fmaxf(m, lsAlpha[tid][i]);
      lsM[tid] = m;
    }
    __syncthreads();

    // ea for next step (f16, in (0,1])
    for (int e = tid; e < NB * KK; e += 256) {
      const int bb = e >> 7, j = e & (KK - 1);
      lsEA[bb][j] = (_Float16)__expf(lsAlpha[bb][j] - lsM[bb]);
    }
    __syncthreads();
  }

  // termination: fwd[b] = logsumexp_i(alpha[i,b] + trans[END,i]); loss = fwd - gold
  if (tid < NB) {
    const int b = tid;
    float m = -1e30f;
    for (int i = 0; i < KK; ++i)
      m = fmaxf(m, lsAlpha[b][i] + trans[ENDD * KK + i]);
    float s = 0.f;
    for (int i = 0; i < KK; ++i)
      s += __expf(lsAlpha[b][i] + trans[ENDD * KK + i] - m);
    loss_out[b0 + b] = (m + __logf(s)) - gold[b0 + b];
  }
}

// ---------------------------------------------------------------------------
// Viterbi DP (max-plus, VALU): one workgroup per batch, one lane per tag i.
// transitions resident in padded LDS; byte backpointers to global (L2).
// ---------------------------------------------------------------------------
__launch_bounds__(128)
__global__ void crf_viterbi(const float* __restrict__ feats,
                            const float* __restrict__ trans,
                            unsigned char* __restrict__ bp,
                            int* __restrict__ bestIdx,
                            float* __restrict__ score_out) {
  __shared__ float lsT[KK][KK + 1];   // padded: conflict-free row reads
  __shared__ float lsS[2][KK];
  __shared__ float red[KK];
  __shared__ int   redi[KK];

  const int i = threadIdx.x;
  const int b = blockIdx.x;

  for (int idx = i; idx < KK * KK; idx += KK)
    lsT[idx >> 7][idx & (KK - 1)] = trans[idx];
  lsS[0][i] = feats[((size_t)b * TT) * KK + i] + trans[i * KK + STARTT];
  __syncthreads();

  int cur = 0;
  for (int t = 1; t < TT; ++t) {
    const float f = feats[((size_t)b * TT + t) * KK + i];
    float best = -1e30f;
    int bj = 0;
#pragma unroll 4
    for (int j = 0; j < KK; ++j) {
      const float v = lsS[cur][j] + lsT[i][j];   // lsS broadcast, lsT conflict-free
      if (v > best) { best = v; bj = j; }        // strict '>' == first-argmax
    }
    bp[((size_t)t * BB + b) * KK + i] = (unsigned char)bj;
    lsS[cur ^ 1][i] = f + best;
    cur ^= 1;
    __syncthreads();
  }

  // terminal argmax over i of score + trans[END,i]
  red[i]  = lsS[cur][i] + trans[ENDD * KK + i];
  redi[i] = i;
  __syncthreads();
  for (int off = 64; off > 0; off >>= 1) {
    if (i < off) {
      if (red[i + off] > red[i]) { red[i] = red[i + off]; redi[i] = redi[i + off]; }
    }
    __syncthreads();
  }
  if (i == 0) { bestIdx[b] = redi[0]; score_out[b] = red[0]; }
}

// ---------------------------------------------------------------------------
// Backtrace: one lane per batch walking byte backpointers (L2-resident).
// ---------------------------------------------------------------------------
__global__ void crf_backtrace(const unsigned char* __restrict__ bp,
                              const int* __restrict__ bestIdx,
                              float* __restrict__ path_out) {
  const int b = blockIdx.x * blockDim.x + threadIdx.x;
  if (b >= BB) return;
  int idx = bestIdx[b];
  path_out[(size_t)b * TT + (TT - 1)] = (float)idx;
  for (int t = TT - 1; t >= 1; --t) {
    idx = bp[((size_t)t * BB + b) * KK + idx];
    path_out[(size_t)b * TT + (t - 1)] = (float)idx;
  }
}

// ---------------------------------------------------------------------------
extern "C" void kernel_launch(void* const* d_in, const int* in_sizes, int n_in,
                              void* d_out, int out_size, void* d_ws, size_t ws_size,
                              hipStream_t stream) {
  const float* feats = (const float*)d_in[0];   // [B,T,K] f32
  const float* trans = (const float*)d_in[1];   // [K,K]   f32
  const int*   tags  = (const int*)d_in[2];     // [B,T]   i32

  float* out = (float*)d_out;
  float* loss   = out;                 // [B]
  float* path   = out + BB;            // [B,T] (int path written as float)
  float* pscore = out + BB + BB * TT;  // [B]

  char* ws = (char*)d_ws;
  _Float16*      expT    = (_Float16*)(ws + WS_EXPT);
  float*         rowmax  = (float*)(ws + WS_ROWMAX);
  float*         gold    = (float*)(ws + WS_GOLD);
  int*           bestIdx = (int*)(ws + WS_BEST);
  unsigned char* bp      = (unsigned char*)(ws + WS_BP);

  crf_prep     <<<1,       KK,  0, stream>>>(trans, expT, rowmax);
  crf_gold     <<<1,       BB,  0, stream>>>(feats, trans, tags, gold);
  crf_forward  <<<BB / NB, 256, 0, stream>>>(feats, trans, expT, rowmax, gold, loss);
  crf_viterbi  <<<BB,      KK,  0, stream>>>(feats, trans, bp, bestIdx, pscore);
  crf_backtrace<<<1,       BB,  0, stream>>>(bp, bestIdx, path);
}
